// SwinV2B_11501922419304
// MI455X (gfx1250) — matrix operand order repaired and checked by reading, not compiled, not run on hardware
//
#include <hip/hip_runtime.h>
#include <hip/hip_bf16.h>

typedef __attribute__((ext_vector_type(16))) _Float16 v16h;
typedef __attribute__((ext_vector_type(8)))  _Float16 v8h;
typedef __attribute__((ext_vector_type(8)))  float    v8f;
typedef __attribute__((ext_vector_type(4)))  float    f4;

// ---------- helpers ----------
__device__ __forceinline__ v8f wmma_f16(v16h a, v16h b, v8f c) {
  // D = A(16x32) * B(32x16) + C  -> v_wmma_f32_16x16x32_f16
  return __builtin_amdgcn_wmma_f32_16x16x32_f16(
      /*neg_a=*/false, a, /*neg_b=*/false, b,
      /*c_mod=*/(short)0, c, /*reuse_a=*/false, /*reuse_b=*/false);
}

__device__ __forceinline__ v16h cat16(v8h lo, v8h hi) {
  return __builtin_shufflevector(lo, hi, 0,1,2,3,4,5,6,7,8,9,10,11,12,13,14,15);
}

__device__ __forceinline__ v8h ld8h(const _Float16* p) { return *(const v8h*)p; }

__device__ __forceinline__ v8h cvt8(const float* p) {
  f4 a = *(const f4*)p;
  f4 b = *(const f4*)(p + 4);
  v8h r;
  r[0] = (_Float16)a[0]; r[1] = (_Float16)a[1]; r[2] = (_Float16)a[2]; r[3] = (_Float16)a[3];
  r[4] = (_Float16)b[0]; r[5] = (_Float16)b[1]; r[6] = (_Float16)b[2]; r[7] = (_Float16)b[3];
  return r;
}

// ---------- constants ----------
#define CCH   512      // channels
#define HEADS 16
#define DHEAD 32
#define NTOK  64       // tokens per window
#define NWIN  16       // windows per image (mask groups)
#define CH    256      // windows per chunk
#define NCHUNK 8       // 2048 / 256

// ---------- prep: f16 weights + fused qkv bias ----------
__global__ void swin_prep(const float* __restrict__ qkv_w, const float* __restrict__ proj_w,
                          const float* __restrict__ q_bias, const float* __restrict__ v_bias,
                          _Float16* __restrict__ wq, _Float16* __restrict__ wp,
                          float* __restrict__ qkvb) {
  int i = blockIdx.x * blockDim.x + threadIdx.x;
  if (i < 3 * CCH * CCH) wq[i] = (_Float16)qkv_w[i];
  if (i < CCH * CCH)     wp[i] = (_Float16)proj_w[i];
  if (i < 3 * CCH) {
    float b = (i < CCH) ? q_bias[i] : (i < 2 * CCH ? 0.0f : v_bias[i - 2 * CCH]);
    qkvb[i] = b;
  }
}

// ---------- prep: CPB MLP table [(2W-1)^2=225, HEADS] ----------
__global__ __launch_bounds__(512)
void swin_cpb(const float* __restrict__ coords /*[225][2]*/,
              const float* __restrict__ w1 /*[512][2]*/, const float* __restrict__ b1 /*[512]*/,
              const float* __restrict__ w2 /*[16][512]*/, float* __restrict__ tbl /*[225][16]*/) {
  __shared__ float hs[512];
  int i = blockIdx.x;         // table row 0..224
  int t = threadIdx.x;        // hidden unit 0..511
  float c0 = coords[i * 2 + 0], c1 = coords[i * 2 + 1];
  float v = c0 * w1[t * 2 + 0] + c1 * w1[t * 2 + 1] + b1[t];
  hs[t] = fmaxf(v, 0.0f);
  __syncthreads();
  if (t < HEADS) {
    float s = 0.0f;
    for (int j = 0; j < 512; ++j) s += hs[j] * w2[t * 512 + j];
    tbl[i * HEADS + t] = s;
  }
}

// ---------- prep: combined 16*sigmoid(bias) + mask -> [w][h][64][64] ----------
__global__ void swin_bm(const float* __restrict__ tbl, const int* __restrict__ rpi,
                        const float* __restrict__ mask, float* __restrict__ bm) {
  int idx = blockIdx.x * blockDim.x + threadIdx.x;   // 0 .. 16*16*4096-1
  if (idx >= NWIN * HEADS * NTOK * NTOK) return;
  int pos = idx & (NTOK * NTOK - 1);
  int h   = (idx >> 12) & (HEADS - 1);
  int w   = idx >> 16;
  float b = tbl[rpi[pos] * HEADS + h];
  bm[idx] = 16.0f / (1.0f + __expf(-b)) + mask[w * (NTOK * NTOK) + pos];
}

// ---------- GEMM 1: qkv = x @ W^T + bias, scattered per-head f16 ----------
// block = 256 thr (8 waves). wave tile 32(M) x 64(N); block tile 64(M) x 256(N).
__global__ __launch_bounds__(256)
void swin_gemm_qkv(const float* __restrict__ x,       // [CH*64][512] (chunk slice)
                   const _Float16* __restrict__ wq,   // [1536][512]
                   const float* __restrict__ qkvb,    // [1536]
                   _Float16* __restrict__ qc,         // [CH*16][64][32]
                   _Float16* __restrict__ kc,         // [CH*16][64][32]
                   _Float16* __restrict__ vc) {       // [CH*16][32][64] (transposed)
  const int lane = threadIdx.x & 31, wv = threadIdx.x >> 5;
  const int wm = wv >> 2, wn = wv & 3;
  const int g = lane >> 4, lr = lane & 15;
  const int m0 = blockIdx.y * 64 + wm * 32;     // chunk-local row base
  const int n0 = blockIdx.x * 256 + wn * 64;    // output column base (0..1535)

  v8f acc[2][4];
  #pragma unroll
  for (int i = 0; i < 2; ++i)
    #pragma unroll
    for (int j = 0; j < 4; ++j) acc[i][j] = (v8f)0.0f;

  const float* xr0 = x + (size_t)(m0 + lr) * CCH;
  const float* xr1 = x + (size_t)(m0 + 16 + lr) * CCH;

  #pragma unroll 2
  for (int kk = 0; kk < CCH; kk += 32) {
    // speculative prefetch: OOB translations are silently dropped
    __builtin_prefetch(xr0 + kk + 64, 0, 1);
    __builtin_prefetch(xr1 + kk + 64, 0, 1);
    v16h a0 = cat16(cvt8(xr0 + kk + g * 8), cvt8(xr0 + kk + 16 + g * 8));
    v16h a1 = cat16(cvt8(xr1 + kk + g * 8), cvt8(xr1 + kk + 16 + g * 8));
    // fetch all four B fragments first so the load waits stagger across WMMAs
    v16h b[4];
    #pragma unroll
    for (int j = 0; j < 4; ++j) {
      const _Float16* wr = wq + (size_t)(n0 + j * 16 + lr) * CCH + kk + g * 8;
      b[j] = cat16(ld8h(wr), ld8h(wr + 16));
    }
    #pragma unroll
    for (int j = 0; j < 4; ++j) {
      acc[0][j] = wmma_f16(a0, b[j], acc[0][j]);
      acc[1][j] = wmma_f16(a1, b[j], acc[1][j]);
    }
  }

  #pragma unroll
  for (int i = 0; i < 2; ++i) {
    #pragma unroll
    for (int j = 0; j < 4; ++j) {
      int c = n0 + j * 16 + lr;                 // 0..1535
      float bias = qkvb[c];
      int s = c >> 9, r = c & 511, h = r >> 5, dd = r & 31;
      _Float16* base = (s == 0) ? qc : (s == 1) ? kc : vc;
      #pragma unroll
      for (int e = 0; e < 8; ++e) {
        int m = m0 + i * 16 + g * 8 + e;        // chunk-local token row
        int lb = m >> 6, n = m & 63;
        size_t idx = ((size_t)(lb * HEADS + h)) * (NTOK * DHEAD) +
                     (s == 2 ? (size_t)(dd * NTOK + n) : (size_t)(n * DHEAD + dd));
        base[idx] = (_Float16)(acc[i][j][e] + bias);
      }
    }
  }
}

// ---------- attention: one block per (window, head), 128 thr = 4 waves ----------
__global__ __launch_bounds__(128)
void swin_attn(const _Float16* __restrict__ qc, const _Float16* __restrict__ kc,
               const _Float16* __restrict__ vc,
               const float* __restrict__ bm,     // [16][H][64][64] combined bias+mask
               const float* __restrict__ lsc,    // [H]
               _Float16* __restrict__ ao) {      // [CH*64][512]
  __shared__ _Float16 qn_s[NTOK * DHEAD];
  __shared__ _Float16 kn_s[NTOK * DHEAD];
  __shared__ _Float16 p_s[NTOK * NTOK];

  const int bh = blockIdx.x;
  const int lb = bh >> 4;           // chunk-local window
  const int h  = bh & 15;
  const _Float16* qp = qc + (size_t)(lb * HEADS + h) * (NTOK * DHEAD);
  const _Float16* kp = kc + (size_t)(lb * HEADS + h) * (NTOK * DHEAD);
  const _Float16* vp = vc + (size_t)(lb * HEADS + h) * (NTOK * DHEAD);

  // phase 1: cosine-normalize q and k rows -> LDS f16 (vectorized b128 traffic)
  {
    int r = threadIdx.x;            // 0..127 -> 64 q rows + 64 k rows
    const _Float16* src = (r < 64) ? (qp + r * DHEAD) : (kp + (r - 64) * DHEAD);
    _Float16*       dst = (r < 64) ? (qn_s + r * DHEAD) : (kn_s + (r - 64) * DHEAD);
    v8h rv[4];
    #pragma unroll
    for (int u = 0; u < 4; ++u) rv[u] = ld8h(src + u * 8);
    float f[DHEAD]; float ss = 0.0f;
    #pragma unroll
    for (int u = 0; u < 4; ++u)
      #pragma unroll
      for (int i = 0; i < 8; ++i) { float t = (float)rv[u][i]; f[u * 8 + i] = t; ss += t * t; }
    float rs = 1.0f / fmaxf(sqrtf(ss), 1e-12f);
    #pragma unroll
    for (int u = 0; u < 4; ++u) {
      v8h o;
      #pragma unroll
      for (int i = 0; i < 8; ++i) o[i] = (_Float16)(f[u * 8 + i] * rs);
      *(v8h*)(dst + u * 8) = o;
    }
  }
  __syncthreads();

  const int lane = threadIdx.x & 31, wv = threadIdx.x >> 5;
  const int g = lane >> 4, lr = lane & 15;
  const int R0 = wv * 16;                       // query-row strip
  const float scale = __expf(fminf(lsc[h], 4.6051702f));   // exp(min(ls, log 100))

  // phase 2: S = qn . kn^T  (K = 32, one wmma per 16x16 tile)
  v16h aq;
  {
    const _Float16* qr = qn_s + (R0 + lr) * DHEAD;
    aq = cat16(ld8h(qr + g * 8), ld8h(qr + 16 + g * 8));
  }
  v16h kb[4];
  #pragma unroll
  for (int j = 0; j < 4; ++j) {
    const _Float16* kr = kn_s + (j * 16 + lr) * DHEAD + g * 8;
    kb[j] = cat16(ld8h(kr), ld8h(kr + 16));
  }
  v8f sacc[4];
  #pragma unroll
  for (int j = 0; j < 4; ++j) sacc[j] = wmma_f16(aq, kb[j], (v8f)0.0f);

  // softmax over 64 keys; row n lives in VGPR e across the 16 lanes of one group
  const int wmask = lb & (NWIN - 1);
  const float* mbrow = bm + ((size_t)(wmask * HEADS + h)) * (NTOK * NTOK);
  #pragma unroll
  for (int e = 0; e < 8; ++e) {
    int n = R0 + g * 8 + e;                     // query row
    float t4[4]; float mx = -1e30f;
    #pragma unroll
    for (int j = 0; j < 4; ++j) {
      int m = j * 16 + lr;                      // key col
      float t = sacc[j][e] * scale + mbrow[n * NTOK + m];
      t4[j] = t; mx = fmaxf(mx, t);
    }
    #pragma unroll
    for (int s = 1; s < 16; s <<= 1) mx = fmaxf(mx, __shfl_xor(mx, s, 32));
    float sum = 0.0f;
    #pragma unroll
    for (int j = 0; j < 4; ++j) { t4[j] = __expf(t4[j] - mx); sum += t4[j]; }
    #pragma unroll
    for (int s = 1; s < 16; s <<= 1) sum += __shfl_xor(sum, s, 32);
    float inv = 1.0f / sum;
    #pragma unroll
    for (int j = 0; j < 4; ++j) p_s[n * NTOK + j * 16 + lr] = (_Float16)(t4[j] * inv);
  }
  __syncthreads();

  // phase 3: O = P . V  (K = 64 -> two wmma steps; V stored [d][key])
  v8f oacc[2] = {(v8f)0.0f, (v8f)0.0f};
  #pragma unroll
  for (int kk = 0; kk < NTOK; kk += 32) {
    const _Float16* pr = p_s + (R0 + lr) * NTOK + kk;
    v16h pa = cat16(ld8h(pr + g * 8), ld8h(pr + 16 + g * 8));
    v16h vb[2];
    #pragma unroll
    for (int nt = 0; nt < 2; ++nt) {
      const _Float16* vr = vp + (nt * 16 + lr) * NTOK + kk + g * 8;
      vb[nt] = cat16(ld8h(vr), ld8h(vr + 16));
    }
    #pragma unroll
    for (int nt = 0; nt < 2; ++nt) oacc[nt] = wmma_f16(pa, vb[nt], oacc[nt]);
  }
  #pragma unroll
  for (int nt = 0; nt < 2; ++nt) {
    int col = h * DHEAD + nt * 16 + lr;
    #pragma unroll
    for (int e = 0; e < 8; ++e) {
      int n = R0 + g * 8 + e;
      ao[(size_t)(lb * NTOK + n) * CCH + col] = (_Float16)oacc[nt][e];
    }
  }
}

// ---------- GEMM 2: out = ao @ proj_w^T + proj_b (fp32 out) ----------
__global__ __launch_bounds__(256)
void swin_gemm_proj(const _Float16* __restrict__ a,   // [CH*64][512]
                    const _Float16* __restrict__ wp,  // [512][512]
                    const float* __restrict__ pb,     // [512]
                    float* __restrict__ out) {        // [CH*64][512] (chunk slice)
  const int lane = threadIdx.x & 31, wv = threadIdx.x >> 5;
  const int wm = wv >> 2, wn = wv & 3;
  const int g = lane >> 4, lr = lane & 15;
  const int m0 = blockIdx.y * 64 + wm * 32;
  const int n0 = blockIdx.x * 256 + wn * 64;

  v8f acc[2][4];
  #pragma unroll
  for (int i = 0; i < 2; ++i)
    #pragma unroll
    for (int j = 0; j < 4; ++j) acc[i][j] = (v8f)0.0f;

  const _Float16* ar0 = a + (size_t)(m0 + lr) * CCH;
  const _Float16* ar1 = a + (size_t)(m0 + 16 + lr) * CCH;

  #pragma unroll 2
  for (int kk = 0; kk < CCH; kk += 32) {
    __builtin_prefetch(ar0 + kk + 64, 0, 1);
    __builtin_prefetch(ar1 + kk + 64, 0, 1);
    v16h a0 = cat16(ld8h(ar0 + kk + g * 8), ld8h(ar0 + kk + 16 + g * 8));
    v16h a1 = cat16(ld8h(ar1 + kk + g * 8), ld8h(ar1 + kk + 16 + g * 8));
    v16h b[4];
    #pragma unroll
    for (int j = 0; j < 4; ++j) {
      const _Float16* wr = wp + (size_t)(n0 + j * 16 + lr) * CCH + kk + g * 8;
      b[j] = cat16(ld8h(wr), ld8h(wr + 16));
    }
    #pragma unroll
    for (int j = 0; j < 4; ++j) {
      acc[0][j] = wmma_f16(a0, b[j], acc[0][j]);
      acc[1][j] = wmma_f16(a1, b[j], acc[1][j]);
    }
  }

  #pragma unroll
  for (int i = 0; i < 2; ++i) {
    #pragma unroll
    for (int j = 0; j < 4; ++j) {
      int c = n0 + j * 16 + lr;
      float bias = pb[c];
      #pragma unroll
      for (int e = 0; e < 8; ++e) {
        int m = m0 + i * 16 + g * 8 + e;
        out[(size_t)m * CCH + c] = acc[i][j][e] + bias;
      }
    }
  }
}

// ---------- host-side launch ----------
extern "C" void kernel_launch(void* const* d_in, const int* in_sizes, int n_in,
                              void* d_out, int out_size, void* d_ws, size_t ws_size,
                              hipStream_t stream) {
  (void)in_sizes; (void)n_in; (void)out_size; (void)ws_size;
  const float* x       = (const float*)d_in[0];   // [2048,64,512]
  const float* mask    = (const float*)d_in[1];   // [16,64,64]
  const float* qkv_w   = (const float*)d_in[2];   // [1536,512]
  const float* q_bias  = (const float*)d_in[3];   // [512]
  const float* v_bias  = (const float*)d_in[4];   // [512]
  const float* lsc     = (const float*)d_in[5];   // [16,1,1]
  const float* cpb_w1  = (const float*)d_in[6];   // [512,2]
  const float* cpb_b1  = (const float*)d_in[7];   // [512]
  const float* cpb_w2  = (const float*)d_in[8];   // [16,512]
  const float* proj_w  = (const float*)d_in[9];   // [512,512]
  const float* proj_b  = (const float*)d_in[10];  // [512]
  const float* coords  = (const float*)d_in[11];  // [225,2]
  const int*   rpi     = (const int*)d_in[12];    // [64,64]
  float* out = (float*)d_out;

  char* ws = (char*)d_ws;
  size_t off = 0;
  auto take = [&](size_t bytes) -> void* {
    void* p = (void*)(ws + off);
    off = (off + bytes + 255) & ~(size_t)255;
    return p;
  };
  _Float16* wq   = (_Float16*)take((size_t)3 * CCH * CCH * 2);
  _Float16* wp   = (_Float16*)take((size_t)CCH * CCH * 2);
  float*    qkvb = (float*)take((size_t)3 * CCH * 4);
  float*    tbl  = (float*)take((size_t)225 * HEADS * 4);
  float*    bm   = (float*)take((size_t)NWIN * HEADS * NTOK * NTOK * 4);
  const size_t qkv_per = (size_t)CH * HEADS * NTOK * DHEAD;   // halves per tensor
  _Float16* qc = (_Float16*)take(qkv_per * 2);
  _Float16* kc = (_Float16*)take(qkv_per * 2);
  _Float16* vc = (_Float16*)take(qkv_per * 2);
  _Float16* ao = (_Float16*)take((size_t)CH * NTOK * CCH * 2);

  // one-time prep (re-done every call: deterministic)
  swin_prep<<<(3 * CCH * CCH + 255) / 256, 256, 0, stream>>>(qkv_w, proj_w, q_bias, v_bias,
                                                             wq, wp, qkvb);
  swin_cpb<<<225, 512, 0, stream>>>(coords, cpb_w1, cpb_b1, cpb_w2, tbl);
  swin_bm<<<(NWIN * HEADS * NTOK * NTOK + 255) / 256, 256, 0, stream>>>(tbl, rpi, mask, bm);

  for (int ch = 0; ch < NCHUNK; ++ch) {
    const float* xc = x + (size_t)ch * CH * NTOK * CCH;
    float*       oc = out + (size_t)ch * CH * NTOK * CCH;
    swin_gemm_qkv<<<dim3(6, (CH * NTOK) / 64), 256, 0, stream>>>(xc, wq, qkvb, qc, kc, vc);
    swin_attn<<<CH * HEADS, 128, 0, stream>>>(qc, kc, vc, bm, lsc, ao);
    swin_gemm_proj<<<dim3(2, (CH * NTOK) / 64), 256, 0, stream>>>(ao, wp, proj_b, oc);
  }
}
